// LR_CRvNN_32341103738885
// MI455X (gfx1250) — matrix operand order repaired and checked by reading, not compiled, not run on hardware
//
#include <hip/hip_runtime.h>
#include <hip/hip_bf16.h>

#define NB 64      // batch N
#define SS 512     // sequence length
#define DD 512     // hidden dim D
#define CC 1024    // intermediate dim C

typedef __attribute__((ext_vector_type(16))) __bf16 v16bf;
typedef __attribute__((ext_vector_type(8)))  float  v8f;
typedef __attribute__((ext_vector_type(8)))  int    v8i;

union V8F { v8f v; float f[8]; };

__device__ __forceinline__ v16bf frag_load(const unsigned short* p) {
  v8i r = *(const v8i*)p;                 // 32B contiguous per lane
  return __builtin_bit_cast(v16bf, r);
}

__device__ __forceinline__ v8f wmma_bf16(v8f acc, const unsigned short* ap,
                                         const unsigned short* bp) {
  // D = A*B + C, fp32 accumulate, bf16 operands (CDNA5 wave32 WMMA)
  return __builtin_amdgcn_wmma_f32_16x16x32_bf16(
      false, frag_load(ap), false, frag_load(bp), (short)0, acc, false, false);
}

__device__ __forceinline__ unsigned short f2bf(float f) {  // round-to-nearest-even
  unsigned u = __float_as_uint(f);
  u += 0x7FFFu + ((u >> 16) & 1u);
  return (unsigned short)(u >> 16);
}

__device__ __forceinline__ float bsig(float x) { return 1.0f / (1.0f + __expf(-x)); }
__device__ __forceinline__ float bgelu(float x) {
  float x3 = x * x * x;
  return 0.5f * x * (1.0f + tanhf(0.7978845608f * (x + 0.044715f * x3)));
}

// ---- A-fragment index helpers (16-bit A 16x32 layout, ISA 7.12.2) ----
// within a 32-wide K tile: lanes 0-15 hold K {0..7,16..23}, lanes 16-31 hold {8..15,24..31}
__device__ __forceinline__ void a_map(int r32, int& half, int& e) {
  half = (r32 >> 3) & 1;
  e    = (r32 & 7) + ((r32 >> 4) << 3);
}

// ================= weight pack: row-major fp32 (K x N) -> bf16 B-fragments =================
// flat layout: ((nt*KT + kt)*32 + lane)*16 + e ; k=kt*32+8*(lane/16)+(e&7)+16*(e>>3)
__global__ __launch_bounds__(256) void k_pack_w(const float* __restrict__ W,
                                                unsigned short* __restrict__ P,
                                                int K, int Ncols) {
  int KT = K >> 5;
  long long idx = (long long)blockIdx.x * 256 + threadIdx.x;
  if (idx >= (long long)K * Ncols) return;
  int e    = idx & 15;
  int lane = (int)((idx >> 4) & 31);
  long long tile = idx >> 9;
  int kt = (int)(tile % KT);
  int nt = (int)(tile / KT);
  int k = kt * 32 + ((lane >> 4) << 3) + (e & 7) + ((e >> 3) << 4);
  int n = nt * 16 + (lane & 15);
  P[idx] = f2bf(W[(long long)k * Ncols + n]);
}

// ================= pack masked sequence into A-fragments (rows = tokens, K = D) =============
__global__ __launch_bounds__(256) void k_pack_seq(const float* __restrict__ seq,
                                                  const float* __restrict__ mask,
                                                  unsigned short* __restrict__ P) {
  long long idx = (long long)blockIdx.x * 256 + threadIdx.x;  // over (N*S)*D
  int d   = (int)(idx & 511);
  long long row = idx >> 9;                                    // token = n*SS + s
  float v = seq[idx] * mask[row];
  long long mt = row >> 4; int mrow = (int)(row & 15);
  int kt = d >> 5, r32 = d & 31, half, e; a_map(r32, half, e);
  P[(((mt * 16 + kt) * 32) + mrow + half * 16) * 16 + e] = f2bf(v);
}

// ================= prologue GEMM: (N*S x 512) @ Winit(512x512) + b, tiles 2048x32 ==========
__global__ __launch_bounds__(256) void k_gemm_init(const unsigned short* __restrict__ Ap,
                                                   const unsigned short* __restrict__ Bp,
                                                   const float* __restrict__ bias,
                                                   float* __restrict__ outp) {
  int lane = threadIdx.x & 31;
  int gwave = blockIdx.x * 8 + (threadIdx.x >> 5);
  int mt = gwave >> 5;   // NT = 32
  int nt = gwave & 31;
  V8F a0, a1;
#pragma unroll
  for (int r = 0; r < 8; ++r) { a0.f[r] = 0.f; a1.f[r] = 0.f; }
#pragma unroll 2
  for (int kt = 0; kt < 16; kt += 2) {     // K = 512, dual accumulators
    const unsigned short* ap0 = Ap + (((long long)mt * 16 + kt) * 32 + lane) * 16;
    const unsigned short* bp0 = Bp + (((long long)nt * 16 + kt) * 32 + lane) * 16;
    if (kt + 2 < 16) __builtin_prefetch(bp0 + 1024, 0, 3);
    a0.v = wmma_bf16(a0.v, ap0, bp0);
    a1.v = wmma_bf16(a1.v, ap0 + 512, bp0 + 512);
  }
  int col = nt * 16 + (lane & 15);
  int mbase = mt * 16 + ((lane >> 4) << 3);
  float bs = bias[col];
#pragma unroll
  for (int r = 0; r < 8; ++r)
    outp[(long long)(mbase + r) * DD + col] = a0.f[r] + a1.f[r] + bs;
}

// ================= prologue LN: in-place LN*mask, also emit bf16 A-fragments per step =======
__global__ __launch_bounds__(256) void k_ln_init(float* __restrict__ X,
                                                 const float* __restrict__ mask,
                                                 const float* __restrict__ g,
                                                 const float* __restrict__ bb,
                                                 unsigned short* __restrict__ Xp) {
  __shared__ float red[256], red2[256];
  int row = blockIdx.x, tid = threadIdx.x;      // row = n*SS + s
  float v0 = X[(long long)row * DD + tid];
  float v1 = X[(long long)row * DD + tid + 256];
  red[tid] = v0 + v1; red2[tid] = v0 * v0 + v1 * v1;
  __syncthreads();
  for (int o = 128; o > 0; o >>= 1) {
    if (tid < o) { red[tid] += red[tid + o]; red2[tid] += red2[tid + o]; }
    __syncthreads();
  }
  float mean = red[0] * (1.0f / DD);
  float var  = red2[0] * (1.0f / DD) - mean * mean;
  float inv  = rsqrtf(var + 1e-5f);
  float mv   = mask[row];
  int n = row >> 9, s = row & 511;
  int mt = n >> 4, mrow = n & 15;
#pragma unroll
  for (int j = 0; j < 2; ++j) {
    int d = tid + j * 256;
    float v = j ? v1 : v0;
    float r = ((v - mean) * inv * g[d] + bb[d]) * mv;
    X[(long long)row * DD + d] = r;
    int kt = d >> 5, r32 = d & 31, half, e; a_map(r32, half, e);
    Xp[(((((long long)s * 4 + mt) * 16 + kt) * 32) + mrow + half * 16) * 16 + e] = f2bf(r);
  }
}

// ================= init recurrent state + barrier counters ================================
__global__ __launch_bounds__(256) void k_init(const float* __restrict__ START,
                                              float* __restrict__ ht, float* __restrict__ gs,
                                              unsigned short* __restrict__ Hp,
                                              unsigned* __restrict__ bar) {
  int idx = blockIdx.x * 256 + threadIdx.x;    // 0 .. N*D-1
  int n = idx >> 9, d = idx & 511;
  float v = START[d];
  ht[idx] = v; gs[idx] = v;
  int mt = n >> 4, mrow = n & 15;
  int kt = d >> 5, r32 = d & 31, half, e; a_map(r32, half, e);
  Hp[((((long long)mt * 16 + kt) * 32) + mrow + half * 16) * 16 + e] = f2bf(v);
  if (blockIdx.x == 0 && threadIdx.x < 8) bar[threadIdx.x * 64] = 0u;
}

// ================= device-wide barrier (arrive + generation), agent scope =================
__device__ __forceinline__ void gsync(unsigned* cnt, unsigned* gen) {
  __threadfence();
  __syncthreads();
  if (threadIdx.x == 0) {
    unsigned g = __hip_atomic_load(gen, __ATOMIC_RELAXED, __HIP_MEMORY_SCOPE_AGENT);
    unsigned prev = __hip_atomic_fetch_add(cnt, 1u, __ATOMIC_ACQ_REL, __HIP_MEMORY_SCOPE_AGENT);
    if (prev == NB - 1) {
      __hip_atomic_store(cnt, 0u, __ATOMIC_RELAXED, __HIP_MEMORY_SCOPE_AGENT);
      __hip_atomic_fetch_add(gen, 1u, __ATOMIC_RELEASE, __HIP_MEMORY_SCOPE_AGENT);
    } else {
      while (__hip_atomic_load(gen, __ATOMIC_ACQUIRE, __HIP_MEMORY_SCOPE_AGENT) == g)
        __builtin_amdgcn_s_sleep(2);
    }
  }
  __syncthreads();
  __threadfence();
}

// ================= persistent recurrent scan: 64 blocks x 256 threads (512 waves) =========
__global__ __launch_bounds__(256) void k_recurrent(
    const unsigned short* __restrict__ W1p, const unsigned short* __restrict__ W2p,
    const float* __restrict__ b1, const float* __restrict__ b2,
    const float* __restrict__ lng, const float* __restrict__ lnb,
    const float* __restrict__ mask,
    const unsigned short* __restrict__ Xp, const float* __restrict__ Xf,
    unsigned short* __restrict__ Hp, float* __restrict__ ht, float* __restrict__ gs,
    unsigned short* __restrict__ Ip, float* __restrict__ Ct,
    unsigned* __restrict__ bar, float* __restrict__ out) {
  __shared__ float red[256], red2[256];
  const int tid   = threadIdx.x;
  const int lane  = tid & 31;
  const int gwave = blockIdx.x * 8 + (tid >> 5);
  unsigned* cnt = bar;
  unsigned* gen = bar + 64;   // separate cacheline

  for (int t = 0; t < SS; ++t) {
    // -------- Phase A: inter = gelu([ht,x] @ W1 + b1), 4x64 tiles, K=1024 --------
    if (gwave < 256) {
      int mt = gwave >> 6;
      int nt = gwave & 63;
      V8F a0, a1;
#pragma unroll
      for (int r = 0; r < 8; ++r) { a0.f[r] = 0.f; a1.f[r] = 0.f; }
#pragma unroll 2
      for (int kt = 0; kt < 32; kt += 2) {
        const unsigned short* ap0 = (kt < 16)
            ? Hp + (((long long)mt * 16 + kt) * 32 + lane) * 16
            : Xp + (((((long long)t * 4 + mt) * 16 + (kt - 16)) * 32) + lane) * 16;
        const unsigned short* ap1 = (kt + 1 < 16)
            ? Hp + (((long long)mt * 16 + kt + 1) * 32 + lane) * 16
            : Xp + (((((long long)t * 4 + mt) * 16 + (kt + 1 - 16)) * 32) + lane) * 16;
        const unsigned short* bp0 = W1p + (((long long)nt * 32 + kt) * 32 + lane) * 16;
        if (kt + 2 < 32) __builtin_prefetch(bp0 + 1024, 0, 3);
        a0.v = wmma_bf16(a0.v, ap0, bp0);
        a1.v = wmma_bf16(a1.v, ap1, bp0 + 512);
      }
      // fused bias + gelu, store directly in GEMM2 A-fragment layout
      int col  = nt * 16 + (lane & 15);
      float bs = b1[col];
      int mbase = (lane >> 4) << 3;
      int kt2 = col >> 5, r32 = col & 31, half, e; a_map(r32, half, e);
      unsigned short* dst = Ip + (((long long)mt * 32 + kt2) * 32 + half * 16) * 16 + e;
#pragma unroll
      for (int r = 0; r < 8; ++r) {
        int mrow = mbase + r;
        dst[(long long)mrow * 16] = f2bf(bgelu(a0.f[r] + a1.f[r] + bs));
      }
    }
    gsync(cnt, gen);

    // -------- Phase B: contents = inter @ W2 + b2, 4x128 tiles, K=1024 --------
    {
      int mt = gwave >> 7;
      int nt = gwave & 127;
      V8F a0, a1;
#pragma unroll
      for (int r = 0; r < 8; ++r) { a0.f[r] = 0.f; a1.f[r] = 0.f; }
#pragma unroll 2
      for (int kt = 0; kt < 32; kt += 2) {
        const unsigned short* ap0 = Ip  + (((long long)mt * 32 + kt) * 32 + lane) * 16;
        const unsigned short* bp0 = W2p + (((long long)nt * 32 + kt) * 32 + lane) * 16;
        if (kt + 2 < 32) __builtin_prefetch(bp0 + 1024, 0, 3);
        a0.v = wmma_bf16(a0.v, ap0, bp0);
        a1.v = wmma_bf16(a1.v, ap0 + 512, bp0 + 512);
      }
      int col  = nt * 16 + (lane & 15);
      float bs = b2[col];
      int mbase = mt * 16 + ((lane >> 4) << 3);
#pragma unroll
      for (int r = 0; r < 8; ++r)
        Ct[(long long)(mbase + r) * (4 * DD) + col] = a0.f[r] + a1.f[r] + bs;
    }
    gsync(cnt, gen);

    // -------- Phase C: gates + combine + LayerNorm; block n owns batch row n --------
    {
      int n = blockIdx.x;
      const float* c = Ct + (long long)n * (4 * DD);
      float mval = mask[n * SS + t];
      float vals[2];
      float s = 0.f, s2 = 0.f;
#pragma unroll
      for (int j = 0; j < 2; ++j) {
        int d = tid + j * 256;
        float f1 = bsig(c[d]);
        float f2 = bsig(c[DD + d]);
        float ig = bsig(c[2 * DD + d]);
        float pr = c[3 * DD + d];
        float v = f1 * ht[n * DD + d]
                + f2 * Xf[((long long)n * SS + t) * DD + d]
                + ig * pr;
        vals[j] = v; s += v; s2 += v * v;
      }
      red[tid] = s; red2[tid] = s2;
      __syncthreads();
      for (int o = 128; o > 0; o >>= 1) {
        if (tid < o) { red[tid] += red[tid + o]; red2[tid] += red2[tid + o]; }
        __syncthreads();
      }
      float mean = red[0] * (1.0f / DD);
      float var  = red2[0] * (1.0f / DD) - mean * mean;
      float inv  = rsqrtf(var + 1e-5f);
      int mt = n >> 4, mrow = n & 15;
#pragma unroll
      for (int j = 0; j < 2; ++j) {
        int d = tid + j * 256;
        float hn = (vals[j] - mean) * inv * lng[d] + lnb[d];
        ht[n * DD + d] = hn;
        out[((long long)n * SS + t) * DD + d] = hn * mval;
        gs[n * DD + d] = mval * hn + (1.0f - mval) * gs[n * DD + d];
        int kt = d >> 5, r32 = d & 31, half, e; a_map(r32, half, e);
        Hp[((((long long)mt * 16 + kt) * 32) + mrow + half * 16) * 16 + e] = f2bf(hn);
      }
    }
    gsync(cnt, gen);
  }

  // -------- epilogue: global_state to output tail --------
  {
    int n = blockIdx.x;
#pragma unroll
    for (int j = 0; j < 2; ++j) {
      int d = tid + j * 256;
      out[(long long)NB * SS * DD + n * DD + d] = gs[n * DD + d];
    }
  }
}

extern "C" void kernel_launch(void* const* d_in, const int* in_sizes, int n_in,
                              void* d_out, int out_size, void* d_ws, size_t ws_size,
                              hipStream_t stream) {
  (void)in_sizes; (void)n_in; (void)out_size; (void)ws_size;
  const float* seq   = (const float*)d_in[0];
  const float* mask  = (const float*)d_in[1];
  const float* START = (const float*)d_in[2];
  const float* Winit = (const float*)d_in[3];
  const float* binit = (const float*)d_in[4];
  const float* W1    = (const float*)d_in[5];
  const float* b1    = (const float*)d_in[6];
  const float* W2    = (const float*)d_in[7];
  const float* b2    = (const float*)d_in[8];
  const float* lng   = (const float*)d_in[9];
  const float* lnb   = (const float*)d_in[10];
  float* out = (float*)d_out;

  char* base = (char*)d_ws;
  size_t off = 0;
  auto take = [&](size_t bytes) -> void* {
    void* p = base + off;
    off += (bytes + 255) & ~(size_t)255;
    return p;
  };
  unsigned*       bar = (unsigned*)take(1024);
  unsigned short* W1p = (unsigned short*)take((size_t)1024 * 1024 * 2);
  unsigned short* W2p = (unsigned short*)take((size_t)1024 * 2048 * 2);
  unsigned short* Wip = (unsigned short*)take((size_t)512 * 512 * 2);
  unsigned short* SAp = (unsigned short*)take((size_t)NB * SS * DD * 2);  // reused as Xp
  float*          Xf  = (float*)take((size_t)NB * SS * DD * 4);           // pre-LN -> LN'd x
  unsigned short* Hp  = (unsigned short*)take((size_t)NB * DD * 2);
  float*          ht  = (float*)take((size_t)NB * DD * 4);
  float*          gsb = (float*)take((size_t)NB * DD * 4);
  unsigned short* Ip  = (unsigned short*)take((size_t)NB * CC * 2);
  float*          Ct  = (float*)take((size_t)NB * 4 * DD * 4);
  unsigned short* Xp  = SAp;  // SAp is dead after the prologue GEMM

  k_pack_w<<<dim3(1024 * 1024 / 256), 256, 0, stream>>>(W1, W1p, 1024, 1024);
  k_pack_w<<<dim3(1024 * 2048 / 256), 256, 0, stream>>>(W2, W2p, 1024, 2048);
  k_pack_w<<<dim3(512 * 512 / 256),   256, 0, stream>>>(Winit, Wip, 512, 512);
  k_pack_seq<<<dim3(NB * SS * DD / 256), 256, 0, stream>>>(seq, mask, SAp);
  k_gemm_init<<<dim3((2048 * 32) / 8), 256, 0, stream>>>(SAp, Wip, binit, Xf);
  k_ln_init<<<dim3(NB * SS), 256, 0, stream>>>(Xf, mask, lng, lnb, Xp);
  k_init<<<dim3(NB * DD / 256), 256, 0, stream>>>(START, ht, gsb, Hp, bar);
  k_recurrent<<<dim3(NB), 256, 0, stream>>>(W1p, W2p, b1, b2, lng, lnb, mask,
                                            Xp, Xf, Hp, ht, gsb, Ip, Ct, bar, out);
}
